// MessageLayer_14096082665483
// MI455X (gfx1250) — matrix operand order repaired and checked by reading, not compiled, not run on hardware
//
#include <hip/hip_runtime.h>

// ---------------------------------------------------------------------------
// GNN message layer for MI455X (gfx1250, wave32, WMMA f32_16x16x32_f16)
// ---------------------------------------------------------------------------

typedef __attribute__((ext_vector_type(16))) _Float16 v16h;
typedef __attribute__((ext_vector_type(8)))  _Float16 h8;
typedef __attribute__((ext_vector_type(8)))  float    v8f;
typedef __attribute__((ext_vector_type(4)))  float    f32x4;
typedef __attribute__((ext_vector_type(8)))  float    f32x8;

#define DIMN  128          // feature dim
#define WAVES 4            // waves per block
#define TPB   128          // threads per block
#define WS1   272          // LDS row stride (f16 elems) for K=256 weight tiles
#define WS2   144          // LDS row stride for K=128 weight tiles
#define AS    272          // LDS row stride for K=256 activation tiles
#define MS    144          // LDS row stride for K=128 intermediate tile

struct __align__(32) SharedEdge {
  _Float16 w1t[128 * WS1];            // W1^T (K=256 part), [n][k]
  _Float16 w2t[128 * WS2];            // W2^T, [n][k]
  _Float16 atile[WAVES * 16 * AS];    // per-wave gathered [x[row] || x[col]] f16
  _Float16 mtile[WAVES * 16 * MS];    // per-wave intermediate m (f16)
  float b1s[128], g1s[128], be1s[128], b2s[128];
  float w256[128], w257[128];         // W1 rows for edge_attr (rank-2 term)
  float ea[WAVES * 16 * 2];
  int   rows[WAVES * 16];
  int   cols[WAVES * 16];
};
static_assert(sizeof(SharedEdge) <= 320 * 1024, "LDS overflow (edge)");

struct __align__(32) SharedNode {
  _Float16 wut[128 * WS1];            // Wu^T, [n][k], K=256
  _Float16 atile[WAVES * 16 * AS];    // per-wave [x || aggr] f16
  float bus[128], gus[128], beus[128];
};
static_assert(sizeof(SharedNode) <= 320 * 1024, "LDS overflow (node)");

__device__ __forceinline__ void wait_lds() {
  __asm volatile("s_wait_dscnt 0x0" ::: "memory");
}

// Native CDNA5 f32 atomic add, device scope (aggr lives in the 192MB L2).
// No return value -> tracked with STOREcnt; S_ENDPGM's implicit wait drains it.
__device__ __forceinline__ void atomic_add_f32_dev(float* p, float v) {
  asm volatile("global_atomic_add_f32 %0, %1, off scope:SCOPE_DEV"
               :: "v"(p), "v"(v) : "memory");
}

// A-fragment for v_wmma_f32_16x16x32_f16: lane holds row M=lane%16,
// element i -> k = kc*32 + (lane/16)*16 + i  => two contiguous 8xf16 chunks.
__device__ __forceinline__ v16h afrag(const _Float16* row, int kc, int khalf8) {
  h8 lo = *(const h8*)(row + kc * 32 + khalf8 * 2);
  h8 hi = *(const h8*)(row + kc * 32 + 8 + khalf8 * 2);
  return __builtin_shufflevector(lo, hi, 0,1,2,3,4,5,6,7,8,9,10,11,12,13,14,15);
}

// Gather 8 floats -> 8 f16 -> LDS (16B store)
__device__ __forceinline__ void cvt_store8(const float* src, _Float16* dst) {
  f32x4 v0 = ((const f32x4*)src)[0];
  f32x4 v1 = ((const f32x4*)src)[1];
  f32x8 v  = __builtin_shufflevector(v0, v1, 0,1,2,3,4,5,6,7);
  *(h8*)dst = __builtin_convertvector(v, h8);
}

// ---------------------------------------------------------------------------
// Prep: zero aggr
// ---------------------------------------------------------------------------
__global__ void gnn_zero_f32(float* __restrict__ p, int n) {
  int i = blockIdx.x * blockDim.x + threadIdx.x;
  int stride = gridDim.x * blockDim.x;
  for (; i < n; i += stride) p[i] = 0.0f;
}

// Prep: transpose + f32->f16 weights into workspace
//   w1t: [128][256]  w2t: [128][128]  wut: [128][256]   (all [n][k])
__global__ void gnn_prep_weights(const float* __restrict__ w1,
                                 const float* __restrict__ w2,
                                 const float* __restrict__ wu,
                                 _Float16* __restrict__ w1t,
                                 _Float16* __restrict__ w2t,
                                 _Float16* __restrict__ wut) {
  int i = blockIdx.x * blockDim.x + threadIdx.x;
  if (i < 128 * 256) {
    int n = i >> 8, k = i & 255;
    w1t[i] = (_Float16)w1[k * 128 + n];
  } else if (i < 128 * 256 + 128 * 128) {
    int j = i - 128 * 256;
    int n = j >> 7, k = j & 127;
    w2t[j] = (_Float16)w2[k * 128 + n];
  } else if (i < 128 * 256 + 128 * 128 + 128 * 256) {
    int j = i - (128 * 256 + 128 * 128);
    int n = j >> 8, k = j & 255;
    wut[j] = (_Float16)wu[k * 128 + n];
  }
}

// ---------------------------------------------------------------------------
// Edge kernel: m = relu(W2 @ relu(LN(W1 @ [x[row]||x[col]||ea] + b1)) + b2)
//              aggr[row] += m   (native f32 atomic, L2-resident)
// One wave = 16 edges x full N=128.
// ---------------------------------------------------------------------------
__global__ __launch_bounds__(TPB, 1)
void gnn_edge_kernel(const float* __restrict__ x,
                     const long long* __restrict__ eidx,   // [2][E]
                     const float* __restrict__ eattr,      // [E][2]
                     const float* __restrict__ w1,         // [258][128] f32 (rows 256,257 used)
                     const float* __restrict__ b1,
                     const float* __restrict__ g1,
                     const float* __restrict__ be1,
                     const float* __restrict__ b2,
                     const _Float16* __restrict__ w1tG,    // [128][256]
                     const _Float16* __restrict__ w2tG,    // [128][128]
                     float* __restrict__ aggr,
                     int E, int ntiles) {
  __shared__ SharedEdge sm;
  const int tid     = threadIdx.x;
  const int lane    = tid & 31;
  const int wid     = tid >> 5;
  const int laneLow = lane & 15;
  const int laneHi  = lane >> 4;

  // ---- cooperative weight staging (once per block) ----
  {
    const unsigned* src = (const unsigned*)w1tG;
    unsigned* dst = (unsigned*)sm.w1t;
    for (int i = tid; i < 128 * 128; i += TPB) {
      int n = i >> 7, dw = i & 127;
      dst[n * (WS1 / 2) + dw] = src[i];
    }
  }
  {
    const unsigned* src = (const unsigned*)w2tG;
    unsigned* dst = (unsigned*)sm.w2t;
    for (int i = tid; i < 128 * 64; i += TPB) {
      int n = i >> 6, dw = i & 63;
      dst[n * (WS2 / 2) + dw] = src[i];
    }
  }
  for (int i = tid; i < 128; i += TPB) {
    sm.b1s[i]  = b1[i];
    sm.g1s[i]  = g1[i];
    sm.be1s[i] = be1[i];
    sm.b2s[i]  = b2[i];
    sm.w256[i] = w1[256 * 128 + i];
    sm.w257[i] = w1[257 * 128 + i];
  }
  __syncthreads();

  _Float16* myA = sm.atile + wid * 16 * AS;
  _Float16* myM = sm.mtile + wid * 16 * MS;

  for (int tile = blockIdx.x * WAVES + wid; tile < ntiles; tile += gridDim.x * WAVES) {
    const int e0 = tile * 16;

    // ---- load edge indices + edge attrs for this 16-edge tile ----
    {
      int m = laneLow;
      int e = e0 + m; if (e >= E) e = E - 1;
      long long v = eidx[(laneHi ? (long long)E : 0ll) + e];
      if (laneHi) sm.cols[wid * 16 + m] = (int)v;
      else        sm.rows[wid * 16 + m] = (int)v;
      if (!laneHi) {
        sm.ea[(wid * 16 + m) * 2 + 0] = eattr[(size_t)e * 2 + 0];
        sm.ea[(wid * 16 + m) * 2 + 1] = eattr[(size_t)e * 2 + 1];
      }
    }
    wait_lds();

    // ---- gather h = [x[row] || x[col]] -> f16 A-tile ----
    #pragma unroll 4
    for (int m = 0; m < 16; ++m) {
      int src = laneHi ? sm.cols[wid * 16 + m] : sm.rows[wid * 16 + m];
      cvt_store8(x + (size_t)src * DIMN + laneLow * 8, myA + m * AS + lane * 8);
    }
    wait_lds();

    // ---- GEMM1: accumulators seeded with b1 + edge_attr rank-2 term ----
    v8f acc[8];
    {
      float biasv[8], wav[8], wbv[8];
      #pragma unroll
      for (int t = 0; t < 8; ++t) {
        int n = t * 16 + laneLow;
        biasv[t] = sm.b1s[n];
        wav[t]   = sm.w256[n];
        wbv[t]   = sm.w257[n];
      }
      #pragma unroll
      for (int j = 0; j < 8; ++j) {
        int m = j + laneHi * 8;
        float ea0 = sm.ea[(wid * 16 + m) * 2 + 0];
        float ea1 = sm.ea[(wid * 16 + m) * 2 + 1];
        #pragma unroll
        for (int t = 0; t < 8; ++t)
          acc[t][j] = biasv[t] + ea0 * wav[t] + ea1 * wbv[t];
      }
    }
    #pragma unroll
    for (int kc = 0; kc < 8; ++kc) {          // K = 256
      v16h a = afrag(myA + laneLow * AS, kc, laneHi * 8);
      v16h bfr[8];                            // preload all B frags -> one wait, 8 b2b WMMAs
      #pragma unroll
      for (int t = 0; t < 8; ++t)
        bfr[t] = *(const v16h*)(sm.w1t + (t * 16 + laneLow) * WS1 + kc * 32 + laneHi * 16);
      #pragma unroll
      for (int t = 0; t < 8; ++t)
        acc[t] = __builtin_amdgcn_wmma_f32_16x16x32_f16(false, a, false, bfr[t],
                                                        (short)0, acc[t], false, false);
    }

    // ---- LayerNorm (per edge row M) + ReLU -> f16 m-tile ----
    {
      float gv[8], bv[8];
      #pragma unroll
      for (int t = 0; t < 8; ++t) {
        gv[t] = sm.g1s[t * 16 + laneLow];
        bv[t] = sm.be1s[t * 16 + laneLow];
      }
      const float inv = 1.0f / 128.0f;
      #pragma unroll
      for (int j = 0; j < 8; ++j) {
        float s = 0.0f, ss = 0.0f;
        #pragma unroll
        for (int t = 0; t < 8; ++t) { float v = acc[t][j]; s += v; ss += v * v; }
        #pragma unroll
        for (int mk = 1; mk <= 8; mk <<= 1) {   // reduce within 16-lane half (= one row)
          s  += __shfl_xor(s,  mk, 32);
          ss += __shfl_xor(ss, mk, 32);
        }
        float mu  = s * inv;
        float var = ss * inv - mu * mu;
        float rs  = rsqrtf(var + 1e-5f);
        int m = j + laneHi * 8;
        #pragma unroll
        for (int t = 0; t < 8; ++t) {
          float v = (acc[t][j] - mu) * rs * gv[t] + bv[t];
          v = fmaxf(v, 0.0f);
          myM[m * MS + t * 16 + laneLow] = (_Float16)v;
        }
      }
    }
    wait_lds();

    // ---- GEMM2: m @ W2 + b2 ----
    #pragma unroll
    for (int t = 0; t < 8; ++t) {
      float b2v = sm.b2s[t * 16 + laneLow];
      #pragma unroll
      for (int j = 0; j < 8; ++j) acc[t][j] = b2v;
    }
    #pragma unroll
    for (int kc = 0; kc < 4; ++kc) {          // K = 128
      v16h a = afrag(myM + laneLow * MS, kc, laneHi * 8);
      v16h bfr[8];
      #pragma unroll
      for (int t = 0; t < 8; ++t)
        bfr[t] = *(const v16h*)(sm.w2t + (t * 16 + laneLow) * WS2 + kc * 32 + laneHi * 16);
      #pragma unroll
      for (int t = 0; t < 8; ++t)
        acc[t] = __builtin_amdgcn_wmma_f32_16x16x32_f16(false, a, false, bfr[t],
                                                        (short)0, acc[t], false, false);
    }

    // ---- ReLU + native-atomic scatter-add into aggr[row] (L2 resident) ----
    #pragma unroll
    for (int j = 0; j < 8; ++j) {
      int m = j + laneHi * 8;
      int e = e0 + m;
      if (e < E) {
        int r = sm.rows[wid * 16 + m];
        float* dst = aggr + (size_t)r * DIMN + laneLow;
        #pragma unroll
        for (int t = 0; t < 8; ++t) {
          float v = fmaxf(acc[t][j], 0.0f);
          atomic_add_f32_dev(dst + t * 16, v);
        }
      }
    }
  }
}

// ---------------------------------------------------------------------------
// Node kernel: out = x + relu(LN([x || aggr] @ Wu + bu))
// ---------------------------------------------------------------------------
__global__ __launch_bounds__(TPB, 1)
void gnn_node_kernel(const float* __restrict__ x,
                     const float* __restrict__ aggr,
                     const _Float16* __restrict__ wutG,   // [128][256]
                     const float* __restrict__ bu,
                     const float* __restrict__ gu,
                     const float* __restrict__ beu,
                     float* __restrict__ out,
                     int N, int ntiles) {
  __shared__ SharedNode sm;
  const int tid     = threadIdx.x;
  const int lane    = tid & 31;
  const int wid     = tid >> 5;
  const int laneLow = lane & 15;
  const int laneHi  = lane >> 4;

  {
    const unsigned* src = (const unsigned*)wutG;
    unsigned* dst = (unsigned*)sm.wut;
    for (int i = tid; i < 128 * 128; i += TPB) {
      int n = i >> 7, dw = i & 127;
      dst[n * (WS1 / 2) + dw] = src[i];
    }
  }
  for (int i = tid; i < 128; i += TPB) {
    sm.bus[i]  = bu[i];
    sm.gus[i]  = gu[i];
    sm.beus[i] = beu[i];
  }
  __syncthreads();

  _Float16* myA = sm.atile + wid * 16 * AS;

  for (int tile = blockIdx.x * WAVES + wid; tile < ntiles; tile += gridDim.x * WAVES) {
    const int n0 = tile * 16;

    // ---- stage u = [x || aggr] rows -> f16 A-tile ----
    #pragma unroll 4
    for (int m = 0; m < 16; ++m) {
      int nd = n0 + m; if (nd >= N) nd = N - 1;
      const float* src = (laneHi ? aggr : x) + (size_t)nd * DIMN + laneLow * 8;
      cvt_store8(src, myA + m * AS + lane * 8);
    }
    wait_lds();

    v8f acc[8];
    #pragma unroll
    for (int t = 0; t < 8; ++t) {
      float bv0 = sm.bus[t * 16 + laneLow];
      #pragma unroll
      for (int j = 0; j < 8; ++j) acc[t][j] = bv0;
    }
    #pragma unroll
    for (int kc = 0; kc < 8; ++kc) {          // K = 256
      v16h a = afrag(myA + laneLow * AS, kc, laneHi * 8);
      v16h bfr[8];
      #pragma unroll
      for (int t = 0; t < 8; ++t)
        bfr[t] = *(const v16h*)(sm.wut + (t * 16 + laneLow) * WS1 + kc * 32 + laneHi * 16);
      #pragma unroll
      for (int t = 0; t < 8; ++t)
        acc[t] = __builtin_amdgcn_wmma_f32_16x16x32_f16(false, a, false, bfr[t],
                                                        (short)0, acc[t], false, false);
    }

    // ---- LN + ReLU + residual ----
    float gv[8], bv[8];
    #pragma unroll
    for (int t = 0; t < 8; ++t) {
      gv[t] = sm.gus[t * 16 + laneLow];
      bv[t] = sm.beus[t * 16 + laneLow];
    }
    const float inv = 1.0f / 128.0f;
    #pragma unroll
    for (int j = 0; j < 8; ++j) {
      float s = 0.0f, ss = 0.0f;
      #pragma unroll
      for (int t = 0; t < 8; ++t) { float v = acc[t][j]; s += v; ss += v * v; }
      #pragma unroll
      for (int mk = 1; mk <= 8; mk <<= 1) {
        s  += __shfl_xor(s,  mk, 32);
        ss += __shfl_xor(ss, mk, 32);
      }
      float mu  = s * inv;
      float var = ss * inv - mu * mu;
      float rs  = rsqrtf(var + 1e-5f);
      int m  = j + laneHi * 8;
      int nd = n0 + m;
      if (nd < N) {
        #pragma unroll
        for (int t = 0; t < 8; ++t) {
          int n = t * 16 + laneLow;
          float v = (acc[t][j] - mu) * rs * gv[t] + bv[t];
          v = fmaxf(v, 0.0f);
          out[(size_t)nd * DIMN + n] = x[(size_t)nd * DIMN + n] + v;
        }
      }
    }
  }
}

// ---------------------------------------------------------------------------
// Host launcher
// Inputs: 0:x 1:edge_index(i64) 2:edge_attr 3:W1 4:b1 5:g1 6:be1 7:W2 8:b2
//         9:Wu 10:bu 11:gu 12:beu
// ---------------------------------------------------------------------------
extern "C" void kernel_launch(void* const* d_in, const int* in_sizes, int n_in,
                              void* d_out, int out_size, void* d_ws, size_t ws_size,
                              hipStream_t stream) {
  const float*      x     = (const float*)d_in[0];
  const long long*  eidx  = (const long long*)d_in[1];
  const float*      eattr = (const float*)d_in[2];
  const float*      W1    = (const float*)d_in[3];
  const float*      b1    = (const float*)d_in[4];
  const float*      g1    = (const float*)d_in[5];
  const float*      be1   = (const float*)d_in[6];
  const float*      W2    = (const float*)d_in[7];
  const float*      b2    = (const float*)d_in[8];
  const float*      Wu    = (const float*)d_in[9];
  const float*      bu    = (const float*)d_in[10];
  const float*      gu    = (const float*)d_in[11];
  const float*      beu   = (const float*)d_in[12];

  const int N = in_sizes[0] / DIMN;
  const int E = in_sizes[1] / 2;
  float* out = (float*)d_out;

  // Workspace layout
  char* ws = (char*)d_ws;
  size_t aggrB = ((size_t)N * DIMN * sizeof(float) + 255) & ~(size_t)255;
  float*    aggr = (float*)ws;
  _Float16* w1t  = (_Float16*)(ws + aggrB);
  _Float16* w2t  = w1t + 128 * 256;
  _Float16* wut  = w2t + 128 * 128;

  // 1) zero aggr
  {
    int n = N * DIMN;
    int blocks = (n / 4 + 255) / 256; if (blocks > 2048) blocks = 2048;
    gnn_zero_f32<<<blocks, 256, 0, stream>>>(aggr, n);
  }
  // 2) transpose/convert weights to f16
  {
    int total = 128 * 256 + 128 * 128 + 128 * 256;
    gnn_prep_weights<<<(total + 255) / 256, 256, 0, stream>>>(W1, W2, Wu, w1t, w2t, wut);
  }
  // 3) edge MLP + scatter
  {
    int ntiles = (E + 15) / 16;
    int grid = (ntiles + WAVES - 1) / WAVES; if (grid > 2048) grid = 2048;
    gnn_edge_kernel<<<grid, TPB, 0, stream>>>(x, eidx, eattr, W1, b1, g1, be1, b2,
                                              w1t, w2t, aggr, E, ntiles);
  }
  // 4) node update MLP + residual
  {
    int ntiles = (N + 15) / 16;
    int grid = (ntiles + WAVES - 1) / WAVES; if (grid > 2048) grid = 2048;
    gnn_node_kernel<<<grid, TPB, 0, stream>>>(x, aggr, wut, bu, gu, beu, out, N, ntiles);
  }
}
